// MLPNet_50568944943204
// MI455X (gfx1250) — compile-verified, matrix operand order checked by reading, not run on hardware
//
#include <hip/hip_runtime.h>

#define NUM_EDGES_C 640000
#define GNN_IN_C    128
#define D_IN_C      260
#define K_PAD_C     288          // 9 * 32
#define PITCH_C     296          // padded halves per row -> 592B, conflict-free
#define HID_C       256
#define M_WG_C      128          // edges per workgroup tile
#define THREADS_C   256          // 8 waves (wave32)
#define NT_C        (HID_C / 16) // 16 N-tiles
#define KSTEPS_C    (K_PAD_C / 32) // 9
#define NUM_TILES_C (NUM_EDGES_C / M_WG_C) // 5000
#define GRID_C      625          // 5000 / 625 = 8 tiles per workgroup

typedef _Float16 v16h __attribute__((ext_vector_type(16)));
typedef _Float16 v8h  __attribute__((ext_vector_type(8)));
typedef _Float16 v4h  __attribute__((ext_vector_type(4)));
typedef float    v8f  __attribute__((ext_vector_type(8)));
typedef float    v4f  __attribute__((ext_vector_type(4)));

union ABfrag { v16h v; v8h h[2]; };

__global__ __launch_bounds__(THREADS_C, 1)
void gnn_edge_mlp_wmma(const float* __restrict__ x,
                       const long long* __restrict__ edge_index,
                       const float* __restrict__ edge_dist,
                       const float* __restrict__ w1,
                       const float* __restrict__ b1,
                       const float* __restrict__ w2,
                       const float* __restrict__ b2,
                       float* __restrict__ out)
{
    extern __shared__ __align__(16) _Float16 smem[];
    _Float16* w1_lds = smem;                              // HID_C * PITCH_C halves
    _Float16* a_lds  = smem + HID_C * PITCH_C;            // M_WG_C * PITCH_C halves
    float*    b1_lds = (float*)(a_lds + (size_t)M_WG_C * PITCH_C);
    float*    w2_lds = b1_lds + HID_C;

    const int tid  = threadIdx.x;
    const int lane = tid & 31;
    const int wave = tid >> 5;
    const int l15  = lane & 15;
    const int hiH  = (lane >> 4) & 1;   // 0: lanes 0-15, 1: lanes 16-31
    const int koff = hiH * 8;           // K sub-offset per ISA f16 A/B layout

    // ---- One-time: stage w1 (f32 row-major [k][n]) -> LDS f16 column-major [n][PITCH] ----
    {
        const int n = tid; // one column per thread; w1[k*256+n] reads coalesce across threads
        _Float16* dst = w1_lds + n * PITCH_C;
        for (int kb = 0; kb < PITCH_C; kb += 8) {   // 37 vectorized stores per column
            v8h h;
            #pragma unroll
            for (int j = 0; j < 8; ++j) {
                const int k = kb + j;
                const float v = (k < D_IN_C) ? w1[k * HID_C + n] : 0.0f;
                h[j] = (_Float16)v;
            }
            *(v8h*)(dst + kb) = h;
        }
        b1_lds[n] = b1[n];
        w2_lds[n] = w2[n];
    }
    // ---- One-time: zero a_lds (covers the K 260..295 pad, never rewritten per tile) ----
    {
        const v8h z = {};
        for (int i = tid; i < (M_WG_C * PITCH_C) / 8; i += THREADS_C)
            *(v8h*)(a_lds + 8 * i) = z;
    }
    const float b2v = b2[0];

    for (int tile = blockIdx.x; tile < NUM_TILES_C; tile += gridDim.x) {
        __syncthreads(); // previous tile's compute done reading a_lds (also orders staging)

        // ---- Gather + convert 128 edges: vectorized float4 loads -> v8h ds_store_b128 ----
        {
            const int m    = tid >> 1;     // 0..127 (edge row within tile)
            const int half = tid & 1;      // half 0: x[row] -> K[0:128); half 1: x[col] -> K[128:256)
            const int e    = tile * M_WG_C + m;
            const long long src_node = edge_index[2 * e + half];
            const v4f* src = (const v4f*)(x + src_node * GNN_IN_C); // rows 512B-aligned
            _Float16* dst = a_lds + m * PITCH_C + half * 128;
            #pragma unroll
            for (int k = 0; k < 16; ++k) {             // 16 x (8 floats -> 8 halves)
                const v4f a = src[2 * k];
                const v4f b = src[2 * k + 1];
                v8h h;
                #pragma unroll
                for (int j = 0; j < 4; ++j) { h[j] = (_Float16)a[j]; h[4 + j] = (_Float16)b[j]; }
                *(v8h*)(dst + 8 * k) = h;
            }
            if (half == 1) {               // K[256:260) = edge_dist; K[260:296) stays zero
                const v4f d = *(const v4f*)(edge_dist + 4 * e);
                v4h hd;
                #pragma unroll
                for (int j = 0; j < 4; ++j) hd[j] = (_Float16)d[j];
                *(v4h*)(a_lds + m * PITCH_C + 256) = hd;
            }
        }
        __syncthreads();

        // ---- Compute: wave handles 16-edge M-tile; fuse layer 2 out of the accumulator ----
        const _Float16* arow = a_lds + (wave * 16 + l15) * PITCH_C + koff;
        float acc[8];
        #pragma unroll
        for (int r = 0; r < 8; ++r) acc[r] = 0.0f;

        for (int nt = 0; nt < NT_C; ++nt) {
            const int n = nt * 16 + l15;
            const _Float16* bcol = w1_lds + n * PITCH_C + koff;
            v8f C = {};
            #pragma unroll
            for (int ks = 0; ks < KSTEPS_C; ++ks) {
                const int kb = ks * 32;
                ABfrag A, B;
                A.h[0] = *(const v8h*)(arow + kb);        // K {kb+koff .. +7}
                A.h[1] = *(const v8h*)(arow + kb + 16);   // K {kb+koff+16 .. +23}
                B.h[0] = *(const v8h*)(bcol + kb);
                B.h[1] = *(const v8h*)(bcol + kb + 16);
                C = __builtin_amdgcn_wmma_f32_16x16x32_f16(
                        false, A.v, false, B.v, (short)0, C, false, false);
            }
            const float bias = b1_lds[n];
            const float wv   = w2_lds[n];
            #pragma unroll
            for (int r = 0; r < 8; ++r) {
                float h = C[r] + bias;          // h[m][n], m = r + 8*hiH
                h = h > 0.0f ? h : 0.0f;        // ReLU
                acc[r] = fmaf(h, wv, acc[r]);   // layer-2 partial over this N-tile
            }
        }

        // ---- Reduce over the 16 N-lanes in each half-wave, sigmoid, store ----
        #pragma unroll
        for (int r = 0; r < 8; ++r) {
            float v = acc[r];
            v += __shfl_xor(v, 1, 16);
            v += __shfl_xor(v, 2, 16);
            v += __shfl_xor(v, 4, 16);
            v += __shfl_xor(v, 8, 16);
            acc[r] = v;
        }
        if (l15 == 0) {
            const int ebase = tile * M_WG_C + wave * 16 + hiH * 8; // rows r+8*hiH
            #pragma unroll
            for (int r = 0; r < 8; ++r) {
                const float z = acc[r] + b2v;
                out[ebase + r] = 1.0f / (1.0f + __expf(-z));
            }
        }
    }
}

extern "C" void kernel_launch(void* const* d_in, const int* in_sizes, int n_in,
                              void* d_out, int out_size, void* d_ws, size_t ws_size,
                              hipStream_t stream) {
    (void)in_sizes; (void)n_in; (void)d_ws; (void)ws_size; (void)out_size;
    const float*     x  = (const float*)d_in[0];
    const long long* ei = (const long long*)d_in[1];
    const float*     ed = (const float*)d_in[2];
    const float*     w1 = (const float*)d_in[3];
    const float*     b1 = (const float*)d_in[4];
    const float*     w2 = (const float*)d_in[5];
    const float*     b2 = (const float*)d_in[6];
    float* out = (float*)d_out;

    const size_t lds_bytes = (size_t)(HID_C + M_WG_C) * PITCH_C * sizeof(_Float16)
                           + (size_t)2 * HID_C * sizeof(float); // 229,376 B < 320 KB WGP LDS
    (void)hipFuncSetAttribute((const void*)gnn_edge_mlp_wmma,
                              hipFuncAttributeMaxDynamicSharedMemorySize,
                              (int)lds_bytes);

    gnn_edge_mlp_wmma<<<dim3(GRID_C), dim3(THREADS_C), lds_bytes, stream>>>(
        x, ei, ed, w1, b1, w2, b2, out);
}
